// EC_VQVAE_87144886435980
// MI455X (gfx1250) — matrix operand order repaired and checked by reading, not compiled, not run on hardware
//
#include <hip/hip_runtime.h>
#include <hip/hip_bf16.h>

typedef __attribute__((ext_vector_type(16))) _Float16 v16h;
typedef __attribute__((ext_vector_type(8)))  _Float16 v8h;
typedef __attribute__((ext_vector_type(8)))  float    v8f;

#define B_    64
#define T_    512
#define LO_   516     // conv length (T + K-1 cyclic pad, 'same' zero pad)
#define LP_   520     // LO_ + 2 zero pad each side
#define CB_   1024
#define EMB_  128
#define TEMP_ 0.05f
#define NT_   2       // position tiles per conv block (A-fragment reuse)
#define NPOS_ (NT_ * 16)
#define HALO_ (NPOS_ + 4)
#define NSUP_ 17      // ceil(516/32) supertiles per batch

#if defined(__HIP_DEVICE_COMPILE__) && defined(__gfx1250__) && defined(__has_builtin)
#  if __has_builtin(__builtin_amdgcn_tensor_load_to_lds)
#    define HAVE_TDM 1
#  endif
#endif
#ifndef HAVE_TDM
#  define HAVE_TDM 0
#endif

// ---------------- WMMA fragment helpers (K-contiguous source rows) ----------------
// A 16x32 f16 layout: lane(m,h) element e -> K = kc + (e&7) + 8h + 16*(e>>3)
__device__ __forceinline__ v16h frag_a(const _Float16* rowK, int kc, int h) {
  v8h lo = *(const v8h*)(rowK + kc + 8 * h);
  v8h hi = *(const v8h*)(rowK + kc + 8 * h + 16);
  v16h r;
  #pragma unroll
  for (int e = 0; e < 8; ++e) { r[e] = lo[e]; r[e + 8] = hi[e]; }
  return r;
}
// B 32x16 f16 layout: lane(n,h) element e -> K = kc + (e&7) + 8h + 16*(e>>3)
__device__ __forceinline__ v16h frag_b(const _Float16* rowK, int kc, int h) {
  v8h lo = *(const v8h*)(rowK + kc + 8 * h);
  v8h hi = *(const v8h*)(rowK + kc + 8 * h + 16);
  v16h r;
  #pragma unroll
  for (int e = 0; e < 8; ++e) { r[e] = lo[e]; r[e + 8] = hi[e]; }
  return r;
}

// ---------------- small utility kernels ----------------

__global__ void k_init(float* avg, float* sqsum, float* predsum) {
  int i = blockIdx.x * blockDim.x + threadIdx.x;
  if (i < CB_) avg[i] = 0.f;
  if (i == 0) { sqsum[0] = 0.f; predsum[0] = 0.f; }
}

// weights (Co, Ktot) f32 -> (Co, KtotPad) f16, zero-padded K tail
__global__ void k_w2h_pad(const float* __restrict__ s, _Float16* __restrict__ d,
                          int Ktot, int KtotPad, int total) {
  int i = blockIdx.x * blockDim.x + threadIdx.x;
  if (i >= total) return;
  int co = i / KtotPad, k = i % KtotPad;
  d[i] = (k < Ktot) ? (_Float16)s[(size_t)co * Ktot + k] : (_Float16)0.f;
}

// x (B,512,16) -> f16 padded (B,16,520); cyclic pad of first 4 timesteps + zero halo
__global__ void k_pack_x(const float* __restrict__ x, _Float16* __restrict__ o) {
  int i = blockIdx.x * blockDim.x + threadIdx.x;
  int total = B_ * 16 * LP_;
  if (i >= total) return;
  int j = i % LP_; int bc = i / LP_;
  int c = bc % 16; int b = bc / 16;
  float v = 0.f;
  int l = j - 2;
  if (l >= 0 && l < LO_) {
    int t = l & 511;
    v = x[((size_t)b * T_ + t) * 16 + c];
  }
  o[i] = (_Float16)v;
}

// H (B,16,516) -> f16 padded (B,16,520); cyclic pad + zero halo (predictor input)
__global__ void k_pack_h(const float* __restrict__ h, _Float16* __restrict__ o) {
  int i = blockIdx.x * blockDim.x + threadIdx.x;
  int total = B_ * 16 * LP_;
  if (i >= total) return;
  int j = i % LP_; int bc = i / LP_;
  int c = bc % 16; int b = bc / 16;
  float v = 0.f;
  int l = j - 2;
  if (l >= 0 && l < LO_) {
    int t = l & 511;
    v = h[((size_t)b * 16 + c) * LO_ + t];
  }
  o[i] = (_Float16)v;
}

// ---------------- WMMA conv1d (K=5, pad=2) as GEMM ----------------
// in: (B, Ci, LP_) f16 ; Wp: (Co, KtotPad) f16 ; out: (B, Co, LO_) f32
// Block: 32 positions (2 N-tiles, A-fragment reuse) x 128 out-channels (8 waves).
// LDS: Bm[NPOS_][KtotPad] im2col (position-major, K-contiguous) + raw[Ci][HALO_].
__global__ __launch_bounds__(256)
void k_conv(const _Float16* __restrict__ inp, const _Float16* __restrict__ Wp,
            const float* __restrict__ bias, float* __restrict__ out,
            int Ci, int Co) {
  int Ktot = Ci * 5;
  int KtotPad = (Ktot + 31) & ~31;
  int tileL = blockIdx.x % NSUP_;
  int b     = blockIdx.x / NSUP_;
  int l0    = tileL * NPOS_;
  int wave  = threadIdx.x >> 5;
  int lane  = threadIdx.x & 31;
  int co0   = (blockIdx.y * 8 + wave) * 16;

  extern __shared__ __align__(16) _Float16 sh[];
  _Float16* Bm  = sh;                       // NPOS_ * KtotPad
  _Float16* raw = sh + NPOS_ * KtotPad;     // Ci * HALO_

#if HAVE_TDM
  // ---- Tensor Data Mover: 2D tile (Ci rows x HALO_ halves) global -> LDS ----
  if (threadIdx.x < 32) {
    typedef unsigned int u32x4 __attribute__((ext_vector_type(4)));
    typedef int i32x8 __attribute__((ext_vector_type(8)));
    typedef int i32x4 __attribute__((ext_vector_type(4)));
    unsigned lds_off = (unsigned)(size_t)(void*)raw;
    unsigned long long ga =
        (unsigned long long)(size_t)(inp + (size_t)b * Ci * LP_ + l0);
    u32x4 g0;
    g0[0] = 1u;                                        // count=1, user mode, no gather
    g0[1] = lds_off;                                   // lds_addr
    g0[2] = (unsigned)(ga & 0xFFFFFFFFu);              // global_addr[31:0]
    g0[3] = (unsigned)((ga >> 32) & 0x01FFFFFFu) | (2u << 30);  // addr[56:32], type=2
    i32x8 g1;
    g1[0] = (int)(1u << 16);                           // data_size=1 (2 bytes)
    g1[1] = (int)((unsigned)LP_ << 16);                // tensor_dim0[15:0] @ bits63:48
    g1[2] = (int)(((unsigned)LP_ >> 16) | ((unsigned)Ci << 16)); // dim0 hi | dim1 lo
    g1[3] = (int)((((unsigned)Ci >> 16) & 0xFFFFu) | ((unsigned)HALO_ << 16)); // dim1 hi | tile_dim0
    g1[4] = (int)((unsigned)Ci & 0xFFFFu);             // tile_dim1=Ci, tile_dim2=0
    g1[5] = LP_;                                       // tensor_dim0_stride lo32
    g1[6] = 0;
    g1[7] = 0;
    i32x4 z4 = {0, 0, 0, 0};
#if __clang_major__ >= 23
    i32x8 z8 = {0, 0, 0, 0, 0, 0, 0, 0};
    __builtin_amdgcn_tensor_load_to_lds(g0, g1, z4, z4, z8, 0);
#else
    __builtin_amdgcn_tensor_load_to_lds(g0, g1, z4, z4, 0);
#endif
    __builtin_amdgcn_s_wait_tensorcnt(0);
  }
  __syncthreads();
#else
  for (int i = threadIdx.x; i < Ci * HALO_; i += 256) {
    int ci = i / HALO_, j = i % HALO_;
    int l = l0 + j;
    raw[i] = (l < LP_) ? inp[(size_t)b * Ci * LP_ + (size_t)ci * LP_ + l] : (_Float16)0.f;
  }
  __syncthreads();
#endif

  // im2col expand: Bm[n][K] = raw[K/5][n + K%5], K-tail zero
  for (int K = threadIdx.x; K < KtotPad; K += 256) {
    int ci = K / 5, kp = K - ci * 5;
    bool ok = (K < Ktot);
    #pragma unroll
    for (int n = 0; n < NPOS_; ++n)
      Bm[n * KtotPad + K] = ok ? raw[ci * HALO_ + n + kp] : (_Float16)0.f;
  }
  __syncthreads();
  if (co0 >= Co) return;

  int m = lane & 15, h = lane >> 4;
  const _Float16* wrow  = Wp + (size_t)(co0 + m) * KtotPad;
  const _Float16* brow0 = Bm + (size_t)m * KtotPad;
  const _Float16* brow1 = Bm + (size_t)(16 + m) * KtotPad;
  v8f acc0 = {}, acc1 = {};
  for (int kc = 0; kc < KtotPad; kc += 32) {
    __builtin_prefetch(wrow + kc + 128, 0, 1);
    v16h a = frag_a(wrow, kc, h);                    // one A load feeds two WMMAs
    v16h bf0 = frag_b(brow0, kc, h);
    v16h bf1 = frag_b(brow1, kc, h);
    acc0 = __builtin_amdgcn_wmma_f32_16x16x32_f16(false, a, false, bf0,
                                                  (short)0, acc0, false, false);
    acc1 = __builtin_amdgcn_wmma_f32_16x16x32_f16(false, a, false, bf1,
                                                  (short)0, acc1, false, false);
  }
  #pragma unroll
  for (int nt = 0; nt < NT_; ++nt) {
    int l = l0 + nt * 16 + m;
    if (l < LO_) {
      v8f acc = nt ? acc1 : acc0;
      #pragma unroll
      for (int r = 0; r < 8; ++r) {
        int co = co0 + r + 8 * h;
        out[(size_t)b * Co * LO_ + (size_t)co * LO_ + l] = acc[r] + bias[co];
      }
    }
  }
}

// ---------------- BatchNorm (training batch stats over B and L) ----------------

__global__ void k_bn_stats(const float* __restrict__ h, int Co,
                           float* __restrict__ meanb, float* __restrict__ rstdb) {
  int c = blockIdx.x;
  const int N = B_ * LO_;
  float s = 0.f, s2 = 0.f;
  for (int i = threadIdx.x; i < N; i += 256) {
    int b = i / LO_, l = i % LO_;
    float v = h[(size_t)b * Co * LO_ + (size_t)c * LO_ + l];
    s += v; s2 += v * v;
  }
  __shared__ float ls[256], ls2[256];
  ls[threadIdx.x] = s; ls2[threadIdx.x] = s2;
  __syncthreads();
  for (int st = 128; st > 0; st >>= 1) {
    if (threadIdx.x < st) { ls[threadIdx.x] += ls[threadIdx.x + st];
                            ls2[threadIdx.x] += ls2[threadIdx.x + st]; }
    __syncthreads();
  }
  if (threadIdx.x == 0) {
    float mean = ls[0] / N;
    float var  = ls2[0] / N - mean * mean;   // biased
    meanb[c] = mean;
    rstdb[c] = rsqrtf(var + 1e-5f);
  }
}

__global__ void k_bn_relu_pack(const float* __restrict__ h, int Co,
                               const float* __restrict__ gamma, const float* __restrict__ beta,
                               const float* __restrict__ meanb, const float* __restrict__ rstdb,
                               _Float16* __restrict__ o) {
  size_t i = (size_t)blockIdx.x * 256 + threadIdx.x;
  size_t total = (size_t)B_ * Co * LP_;
  if (i >= total) return;
  int j = (int)(i % LP_);
  size_t bc = i / LP_;
  int c = (int)(bc % Co); int b = (int)(bc / Co);
  float v = 0.f;
  int l = j - 2;
  if (l >= 0 && l < LO_) {
    float x = h[(size_t)b * Co * LO_ + (size_t)c * LO_ + l];
    x = gamma[c] * (x - meanb[c]) * rstdb[c] + beta[c];
    v = fmaxf(x, 0.f);
  }
  o[i] = (_Float16)v;
}

// ---------------- codebook prep ----------------

__global__ void k_enorm(const float* __restrict__ E, _Float16* __restrict__ E16,
                        float* __restrict__ enorm) {
  int j = blockIdx.x, d = threadIdx.x;
  float v = E[(size_t)j * EMB_ + d];
  E16[(size_t)j * EMB_ + d] = (_Float16)v;
  __shared__ float ls[EMB_];
  ls[d] = v * v; __syncthreads();
  for (int st = 64; st > 0; st >>= 1) {
    if (d < st) ls[d] += ls[d + st];
    __syncthreads();
  }
  if (d == 0) enorm[j] = ls[0];
}

// ---------------- fused VQ: WMMA distances + argmin + softmax avg + q pack ----------------
__global__ __launch_bounds__(256)
void k_vq(const float* __restrict__ z, const _Float16* __restrict__ E16,
          const float* __restrict__ Ef, const float* __restrict__ enorm,
          int* __restrict__ idx_out, float* __restrict__ idxf_out,
          _Float16* __restrict__ qpad,      // (B,128,520), halo already zero
          float* __restrict__ avg, float* __restrict__ sqsum) {
  __shared__ __align__(16) _Float16 zh[16 * EMB_];
  __shared__ float    zf[16 * EMB_];
  __shared__ float    G[16 * CB_];
  __shared__ float    znorm[16];
  __shared__ float    red[256];
  __shared__ int      redi[256];
  __shared__ float    dmin[16], zsum[16];
  __shared__ int      imin[16];

  int p0 = blockIdx.x * 16;
  int tid = threadIdx.x;

  for (int i = tid; i < 16 * EMB_; i += 256) {
    int r = i >> 7, d = i & 127;
    int p = p0 + r; int b = p >> 9, t = p & 511;
    float v = z[(size_t)b * EMB_ * LO_ + (size_t)d * LO_ + t];
    zf[i] = v; zh[i] = (_Float16)v;
  }
  __syncthreads();
  if (tid < 16) {
    float s = 0.f;
    for (int d = 0; d < EMB_; ++d) { float v = zf[tid * EMB_ + d]; s += v * v; }
    znorm[tid] = s;
  }
  __syncthreads();

  // G = z @ E^T   (M=16 positions, N=1024 codes, K=128)
  int wave = tid >> 5, lane = tid & 31;
  int m = lane & 15, h = lane >> 4;
  const _Float16* arow = zh + m * EMB_;
  for (int s = 0; s < 8; ++s) {
    int j0 = (s * 8 + wave) * 16;
    const _Float16* brow = E16 + (size_t)(j0 + m) * EMB_;
    v8f acc = {};
    #pragma unroll
    for (int kc = 0; kc < EMB_; kc += 32) {
      v16h a  = frag_a(arow, kc, h);
      v16h bf = frag_b(brow, kc, h);
      acc = __builtin_amdgcn_wmma_f32_16x16x32_f16(false, a, false, bf,
                                                   (short)0, acc, false, false);
    }
    #pragma unroll
    for (int r = 0; r < 8; ++r) G[(r + 8 * h) * CB_ + j0 + m] = acc[r];
  }
  __syncthreads();

  // per-row argmin over 1024 codes: d_j = |z|^2 + |e_j|^2 - 2 G
  int r = tid >> 4, sub = tid & 15;
  float best = 3.4e38f; int bi = 0;
  for (int j = sub; j < CB_; j += 16) {
    float d = znorm[r] + enorm[j] - 2.f * G[r * CB_ + j];
    if (d < best) { best = d; bi = j; }
  }
  red[tid] = best; redi[tid] = bi;
  __syncthreads();
  for (int st = 8; st > 0; st >>= 1) {
    if (sub < st && red[tid + st] < red[tid]) { red[tid] = red[tid + st]; redi[tid] = redi[tid + st]; }
    __syncthreads();
  }
  if (sub == 0) { dmin[r] = red[tid]; imin[r] = redi[tid]; }
  __syncthreads();

  float es = 0.f;
  for (int j = sub; j < CB_; j += 16) {
    float d = znorm[r] + enorm[j] - 2.f * G[r * CB_ + j];
    es += __expf((dmin[r] - d) * (1.f / TEMP_));
  }
  red[tid] = es; __syncthreads();
  for (int st = 8; st > 0; st >>= 1) {
    if (sub < st) red[tid] += red[tid + st];
    __syncthreads();
  }
  if (sub == 0) zsum[r] = red[tid];
  __syncthreads();

  float inv = 1.f / (zsum[r] * 32768.f);
  for (int j = sub; j < CB_; j += 16) {
    float d = znorm[r] + enorm[j] - 2.f * G[r * CB_ + j];
    atomicAdd(&avg[j], __expf((dmin[r] - d) * (1.f / TEMP_)) * inv);
  }

  if (tid < 16) {
    int p = p0 + tid;
    idx_out[p] = imin[tid];
    idxf_out[p] = (float)imin[tid];
  }
  __syncthreads();

  float local = 0.f;
  for (int i = tid; i < 16 * EMB_; i += 256) {
    int rr = i >> 7, d = i & 127;
    int p = p0 + rr; int b = p >> 9, t = p & 511;
    float qv = Ef[(size_t)imin[rr] * EMB_ + d];
    _Float16 qh = (_Float16)qv;
    size_t base = (size_t)b * EMB_ * LP_ + (size_t)d * LP_;
    qpad[base + 2 + t] = qh;
    if (t < 4) qpad[base + 2 + 512 + t] = qh;
    float diff = qv - zf[i];
    local += diff * diff;
  }
  red[tid] = local; __syncthreads();
  for (int st = 128; st > 0; st >>= 1) {
    if (tid < st) red[tid] += red[tid + st];
    __syncthreads();
  }
  if (tid == 0) atomicAdd(sqsum, red[0]);
}

// ---------------- outputs & losses ----------------

__global__ void k_xrecon(const float* __restrict__ h, float* __restrict__ o) {
  int i = blockIdx.x * blockDim.x + threadIdx.x;
  if (i >= B_ * T_ * 16) return;
  int c = i & 15; int bt = i >> 4;
  int t = bt % T_; int b = bt / T_;
  o[i] = h[((size_t)b * 16 + c) * LO_ + t];
}

__global__ void k_pred_loss(const float* __restrict__ logits, // (B,1024,516)
                            const int* __restrict__ idx, float* __restrict__ acc) {
  int p = blockIdx.x;
  int b = p >> 9, t = p & 511;
  const float* base = logits + (size_t)b * CB_ * LO_ + t;
  __shared__ float ls[256];
  float mx = -3.4e38f;
  for (int j = threadIdx.x; j < CB_; j += 256) mx = fmaxf(mx, base[(size_t)j * LO_]);
  ls[threadIdx.x] = mx; __syncthreads();
  for (int st = 128; st > 0; st >>= 1) {
    if (threadIdx.x < st) ls[threadIdx.x] = fmaxf(ls[threadIdx.x], ls[threadIdx.x + st]);
    __syncthreads();
  }
  mx = ls[0]; __syncthreads();
  float s = 0.f;
  for (int j = threadIdx.x; j < CB_; j += 256) s += __expf(base[(size_t)j * LO_] - mx);
  ls[threadIdx.x] = s; __syncthreads();
  for (int st = 128; st > 0; st >>= 1) {
    if (threadIdx.x < st) ls[threadIdx.x] += ls[threadIdx.x + st];
    __syncthreads();
  }
  if (threadIdx.x == 0) {
    float lse = mx + __logf(ls[0]);
    float lp = base[(size_t)idx[p] * LO_] - lse;
    atomicAdd(acc, -lp * (1.f / 32768.f));
  }
}

__global__ void k_finalize(const float* __restrict__ avg, const float* __restrict__ sqsum,
                           const float* __restrict__ predsum, float* __restrict__ out3) {
  __shared__ float ls[256];
  float s = 0.f;
  for (int j = threadIdx.x; j < CB_; j += 256) {
    float a = avg[j];
    s += a * __logf(fmaxf(a, 1e-10f));
  }
  ls[threadIdx.x] = s; __syncthreads();
  for (int st = 128; st > 0; st >>= 1) {
    if (threadIdx.x < st) ls[threadIdx.x] += ls[threadIdx.x + st];
    __syncthreads();
  }
  if (threadIdx.x == 0) {
    out3[0] = 1.25f * sqsum[0] * (1.f / (32768.f * 128.f));   // vq_loss
    out3[1] = -ls[0];                                          // entropy
    out3[2] = predsum[0];                                      // pred_loss
  }
}

// ---------------- host launcher ----------------

static inline size_t alignup(size_t x) { return (x + 255) & ~(size_t)255; }

extern "C" void kernel_launch(void* const* d_in, const int* in_sizes, int n_in,
                              void* d_out, int out_size, void* d_ws, size_t ws_size,
                              hipStream_t stream) {
  (void)in_sizes; (void)n_in; (void)out_size; (void)ws_size;
  const float* x = (const float*)d_in[0];
  const float* E = (const float*)d_in[1];
  const float* dW[3] = {(const float*)d_in[2],  (const float*)d_in[6],  (const float*)d_in[10]};
  const float* db[3] = {(const float*)d_in[3],  (const float*)d_in[7],  (const float*)d_in[11]};
  const float* dbe[2] = {(const float*)d_in[4], (const float*)d_in[8]};
  const float* dga[2] = {(const float*)d_in[5], (const float*)d_in[9]};
  const float* eW[3] = {(const float*)d_in[12], (const float*)d_in[16], (const float*)d_in[20]};
  const float* eb[3] = {(const float*)d_in[13], (const float*)d_in[17], (const float*)d_in[21]};
  const float* ebe[2] = {(const float*)d_in[14], (const float*)d_in[18]};
  const float* ega[2] = {(const float*)d_in[15], (const float*)d_in[19]};
  const float* pW[3] = {(const float*)d_in[22], (const float*)d_in[26], (const float*)d_in[30]};
  const float* pb[3] = {(const float*)d_in[23], (const float*)d_in[27], (const float*)d_in[31]};
  const float* pbe[2] = {(const float*)d_in[24], (const float*)d_in[28]};
  const float* pga[2] = {(const float*)d_in[25], (const float*)d_in[29]};

  char* ws = (char*)d_ws;
  size_t off = 0;
  auto grab = [&](size_t bytes) -> char* { char* p = ws + off; off += alignup(bytes); return p; };

  _Float16* f16buf = (_Float16*)grab((size_t)B_ * 256 * LP_ * 2);
  float*    H      = (float*)grab((size_t)B_ * 1024 * LO_ * 4);
  _Float16* E16    = (_Float16*)grab((size_t)CB_ * EMB_ * 2);
  float* enorm   = (float*)grab(CB_ * 4);
  float* meanb   = (float*)grab(1024 * 4);
  float* rstdb   = (float*)grab(1024 * 4);
  float* avg     = (float*)grab(CB_ * 4);
  float* sqsum   = (float*)grab(256);
  float* predsum = (float*)grab(256);
  int*   idxb    = (int*)grab(32768 * 4);

  // padded f16 weights: order dec0,dec1,dec2, enc0,enc1,enc2, pred0,pred1,pred2
  const int wCo[9]   = {256, 256, 16, 256, 256, 128, 256, 256, 1024};
  const int wKt[9]   = {640, 1280, 1280, 80, 1280, 1280, 80, 1280, 1280};
  const float* wsrc[9] = {dW[0], dW[1], dW[2], eW[0], eW[1], eW[2], pW[0], pW[1], pW[2]};
  _Float16* wh[9];
  int wKp[9], wTot[9];
  for (int i = 0; i < 9; ++i) {
    wKp[i] = (wKt[i] + 31) & ~31;
    wTot[i] = wCo[i] * wKp[i];
    wh[i] = (_Float16*)grab((size_t)wTot[i] * 2);
  }

  float* outf = (float*)d_out;
  float* xrec = outf;
  float* scalars = outf + (size_t)B_ * T_ * 16;
  float* idxf = scalars + 3;

  k_init<<<dim3(4), dim3(256), 0, stream>>>(avg, sqsum, predsum);
  for (int i = 0; i < 9; ++i)
    k_w2h_pad<<<dim3((wTot[i] + 255) / 256), dim3(256), 0, stream>>>(
        wsrc[i], wh[i], wKt[i], wKp[i], wTot[i]);

  k_pack_x<<<dim3((B_ * 16 * LP_ + 255) / 256), dim3(256), 0, stream>>>(x, f16buf);

  auto conv = [&](int Ci, int Co, _Float16* w, const float* bias) {
    int KtotPad = (Ci * 5 + 31) & ~31;
    size_t shbytes = (size_t)(NPOS_ * KtotPad + Ci * HALO_) * 2;
    dim3 g(NSUP_ * B_, (Co + 127) / 128);
    k_conv<<<g, dim3(256), shbytes, stream>>>(f16buf, w, bias, H, Ci, Co);
  };
  auto bn = [&](int Co, const float* gamma, const float* beta) {
    k_bn_stats<<<dim3(Co), dim3(256), 0, stream>>>(H, Co, meanb, rstdb);
    size_t tot = (size_t)B_ * Co * LP_;
    k_bn_relu_pack<<<dim3((unsigned)((tot + 255) / 256)), dim3(256), 0, stream>>>(
        H, Co, gamma, beta, meanb, rstdb, f16buf);
  };

  // encoder
  conv(16, 256, wh[3], eb[0]);  bn(256, ega[0], ebe[0]);
  conv(256, 256, wh[4], eb[1]); bn(256, ega[1], ebe[1]);
  conv(256, 128, wh[5], eb[2]);                       // H = z (B,128,516)

  // VQ
  k_enorm<<<dim3(CB_), dim3(EMB_), 0, stream>>>(E, E16, enorm);
  k_vq<<<dim3(2048), dim3(256), 0, stream>>>(H, E16, E, enorm, idxb, idxf, f16buf, avg, sqsum);

  // decoder (single pass; both reference decoder calls are numerically identical)
  conv(128, 256, wh[0], db[0]); bn(256, dga[0], dbe[0]);
  conv(256, 256, wh[1], db[1]); bn(256, dga[1], dbe[1]);
  conv(256, 16, wh[2], db[2]);                        // H = x_recon (B,16,516)
  k_xrecon<<<dim3((B_ * T_ * 16 + 255) / 256), dim3(256), 0, stream>>>(H, xrec);
  k_pack_h<<<dim3((B_ * 16 * LP_ + 255) / 256), dim3(256), 0, stream>>>(H, f16buf);

  // predictor
  conv(16, 256, wh[6], pb[0]);  bn(256, pga[0], pbe[0]);
  conv(256, 256, wh[7], pb[1]); bn(256, pga[1], pbe[1]);
  conv(256, 1024, wh[8], pb[2]);                      // H = logits (B,1024,516)

  k_pred_loss<<<dim3(32768), dim3(256), 0, stream>>>(H, idxb, predsum);
  k_finalize<<<dim3(1), dim3(256), 0, stream>>>(avg, sqsum, predsum, scalars);
}
